// RVAE_85152021610757
// MI455X (gfx1250) — compile-verified, matrix operand order checked by reading, not run on hardware
//
#include <hip/hip_runtime.h>
#include <hip/hip_bf16.h>
#include <stdint.h>

#define Bz 128
#define Lz 2048
#define Fz 32
#define Hz 256
#define Zz 4
#define NWG 4
#define NTHREADS 256
#define LOG2PI 1.8378770664093453f

typedef __bf16 bf16;
typedef __attribute__((ext_vector_type(16))) __bf16 v16bf;
typedef __attribute__((ext_vector_type(8)))  float  v8f;
typedef __attribute__((ext_vector_type(4)))  unsigned int v4u;
typedef __attribute__((ext_vector_type(8)))  int v8i;
typedef __attribute__((ext_vector_type(4)))  int v4i;

#if __has_builtin(__builtin_amdgcn_tensor_load_to_lds)
#define HAVE_TDM 1
#else
#define HAVE_TDM 0
#endif

// ---------------- workspace layout (bytes) ----------------
constexpr size_t FRAG_ELEMS = 8*8*32*16;   // 32768 bf16 = 64KB  (h as WMMA A-fragments)
constexpr size_t DLAY_ELEMS = (size_t)4*8*4*8*32; // 32768 f32 = 128KB (h in D-layout)
constexpr size_t OFF_PART    = 64;         // 5 floats: recon partials[4], kld
constexpr size_t OFF_ENCFRAG = 4096;
constexpr size_t OFF_ENCD    = OFF_ENCFRAG + 2*FRAG_ELEMS*2;
constexpr size_t OFF_PHIFRAG = OFF_ENCD    + 2*DLAY_ELEMS*4;
constexpr size_t OFF_PHID    = OFF_PHIFRAG + 2*FRAG_ELEMS*2;
constexpr size_t OFF_PHIPL   = OFF_PHID    + 2*DLAY_ELEMS*4;
constexpr size_t OFF_THFRAG  = OFF_PHIPL   + (size_t)Bz*Hz*2;
constexpr size_t OFF_THD     = OFF_THFRAG  + 2*FRAG_ELEMS*2;
constexpr size_t OFF_XH      = 2u*1024u*1024u;        // bf16 [L][B][H] = 128MB
constexpr size_t INIT_BYTES  = 2u*1024u*1024u;

constexpr int ENC_LDS_BYTES = (12*8*512 + 12*512 + 128*64)*2 + (2*128*32 + 2*192)*4;
constexpr int DEC_LDS_BYTES = (2*12*8*512 + 8*512 + 128*64)*2
                            + (2*192*Zz + 4*192 + 8*512 + 128*Zz)*4;

// ---------------- device helpers ----------------
__device__ __forceinline__ v8f wmma_bf(v16bf a, v16bf b, v8f c) {
  return __builtin_amdgcn_wmma_f32_16x16x32_bf16(false, a, false, b, (short)0, c, false, false);
}

// K index inside a 16x32 fragment for (lane half, slot)
__device__ __forceinline__ int kmap(int half, int s) {
  return (s & 7) + 8*half + 16*(s >> 3);
}

__device__ __forceinline__ void grid_barrier(unsigned* ctr, unsigned target) {
  __syncthreads();
  if (threadIdx.x == 0) {
    __threadfence();
    __hip_atomic_fetch_add(ctr, 1u, __ATOMIC_ACQ_REL, __HIP_MEMORY_SCOPE_AGENT);
    while (__hip_atomic_load(ctr, __ATOMIC_ACQUIRE, __HIP_MEMORY_SCOPE_AGENT) < target)
      __builtin_amdgcn_s_sleep(2);
  }
  __syncthreads();
}

#if HAVE_TDM
// TDM: load 2D tile [128 rows x 32 f32], row stride L*F elements, into LDS
__device__ __forceinline__ void tdm_load_x_tile(const float* gsrc, const float* ldst) {
  unsigned long long ga = (unsigned long long)(size_t)gsrc;
  v4u g0;
  g0.x = 1u;                                          // count = 1 (valid)
  g0.y = (unsigned)(size_t)ldst;                      // LDS byte address (low 32 bits)
  g0.z = (unsigned)(ga & 0xffffffffull);              // global_addr[31:0]
  g0.w = (unsigned)((ga >> 32) & 0x1ffffffull) | (2u << 30); // addr[56:32] | type=2
  const unsigned ts = (unsigned)(Lz * Fz);            // tensor_dim0 / dim0_stride (elements)
  v8i g1;
  g1[0] = (int)(2u << 16);                            // data_size = 4B, no multicast
  g1[1] = (int)((ts & 0xffffu) << 16);                // tensor_dim0[15:0]
  g1[2] = (int)(((ts >> 16) & 0xffffu) | ((unsigned)Bz << 16)); // dim0 hi | tensor_dim1 lo
  g1[3] = (int)((unsigned)Fz << 16);                  // tensor_dim1 hi=0 | tile_dim0=32
  g1[4] = (int)(Bz);                                  // tile_dim1=128 | tile_dim2=0
  g1[5] = (int)(ts);                                  // tensor_dim0_stride[31:0]
  g1[6] = 0; g1[7] = 0;
  v4i z4 = {0,0,0,0};
  v8i z8 = {0,0,0,0,0,0,0,0};
  __builtin_amdgcn_tensor_load_to_lds(g0, g1, z4, z4, z8, 0);
}
#endif

// One GRUCell step (decoder variant: input dim Z=4 handled by VALU), WMMA on h.
// Per WG: owns 64 hidden units -> 12 N-tiles (r:0-3, z:4-7, n:8-11), K=256 (8 k-tiles).
__device__ __forceinline__ void gru_wmma_step(
    const bf16* __restrict__ WfHH, const float* __restrict__ WihC,
    const float* __restrict__ bI, const float* __restrict__ bH,
    const float* __restrict__ zin,
    const bf16* __restrict__ hfrag_rd, bf16* __restrict__ hfrag_wr,
    const float* __restrict__ hD_rd, float* __restrict__ hD_wr,
    bf16* __restrict__ stag, bf16* __restrict__ plain,
    int chunk, int wave, int lane)
{
  const v8f vzero = {0.f,0.f,0.f,0.f,0.f,0.f,0.f,0.f};
  v8f accRZ[8], accN[4];
#pragma unroll
  for (int i = 0; i < 8; i++) accRZ[i] = vzero;
#pragma unroll
  for (int i = 0; i < 4; i++) accN[i] = vzero;

#pragma unroll 2
  for (int kt = 0; kt < 8; ++kt) {
    v16bf a = *(const v16bf*)(hfrag_rd + ((size_t)((wave*8 + kt)*32 + lane))*16);
#pragma unroll
    for (int nt = 0; nt < 12; ++nt) {
      v16bf bm = *(const v16bf*)(WfHH + ((size_t)((nt*8 + kt)*32 + lane))*16);
      if (nt < 8) accRZ[nt]   = wmma_bf(a, bm, accRZ[nt]);
      else        accN[nt-8]  = wmma_bf(a, bm, accN[nt-8]);
    }
  }

  const int n = lane & 15, half = lane >> 4;
#pragma unroll
  for (int i = 0; i < 4; i++) {
    int cR = i*16 + n, cZ = cR + 64, cN = cR + 128;
#pragma unroll
    for (int r = 0; r < 8; r++) {
      int b = wave*16 + r + 8*half;
      const float* zr = zin + b*Zz;
      float giR = bI[cR], giZ = bI[cZ], giN = bI[cN];
#pragma unroll
      for (int k = 0; k < Zz; k++) {
        float zv = zr[k];
        giR += zv * WihC[cR*Zz + k];
        giZ += zv * WihC[cZ*Zz + k];
        giN += zv * WihC[cN*Zz + k];
      }
      float rg = 1.f/(1.f + __expf(-(giR + accRZ[i][r]   + bH[cR])));
      float zg = 1.f/(1.f + __expf(-(giZ + accRZ[4+i][r] + bH[cZ])));
      float ng = tanhf(giN + rg*(accN[i][r] + bH[cN]));
      size_t doff = ((((size_t)chunk*8 + wave)*4 + i)*8 + r)*32 + lane;
      float hn = (1.f - zg)*ng + zg*hD_rd[doff];
      hD_wr[doff] = hn;
      stag[b*64 + i*16 + n] = (bf16)hn;
    }
  }
  __syncthreads();

  // bounce: D-layout LDS staging -> global A-fragment layout (+ optional plain copy)
  const int tid = threadIdx.x;
#pragma unroll
  for (int rr = 0; rr < 2; rr++) {
    int row = tid*2 + rr;
    int l2 = row & 31, ktl = (row >> 5) & 1, mt = row >> 6;
    int mm = l2 & 15, hf = l2 >> 4;
    v16bf tv;
#pragma unroll
    for (int s = 0; s < 16; s++)
      tv[s] = stag[(mt*16 + mm)*64 + ktl*32 + kmap(hf, s)];
    *(v16bf*)(hfrag_wr + ((size_t)((mt*8 + (chunk*2 + ktl))*32 + l2))*16) = tv;
  }
  if (plain) {
    int b = tid >> 1, seg = tid & 1;
    const v16bf* src = (const v16bf*)(stag + b*64 + seg*32);
    v16bf* dst = (v16bf*)(plain + (size_t)b*Hz + chunk*64 + seg*32);
    dst[0] = src[0]; dst[1] = src[1];
  }
}

// ---------------- encoder: reverse GRU over time ----------------
__global__ void __launch_bounds__(NTHREADS) enc_kernel(
    const float* __restrict__ x, const float* __restrict__ Wih1,
    const float* __restrict__ Whh1, const float* __restrict__ bih1,
    const float* __restrict__ bhh1, char* __restrict__ ws)
{
  extern __shared__ char smem[];
  bf16* WfHH  = (bf16*)smem;                 // 12*8*512 bf16
  bf16* WfIH  = WfHH + 12*8*512;             // 12*512
  bf16* stag  = WfIH + 12*512;               // 128*64
  float* xstag = (float*)(stag + 128*64);    // 2 * 128*32 f32
  float* bI   = xstag + 2*128*32;            // 192
  float* bH   = bI + 192;                    // 192

  const int chunk = blockIdx.x;
  const int tid = threadIdx.x;
  const int wave = tid >> 5, lane = tid & 31;

  unsigned* ctr = (unsigned*)ws;
  bf16* hfrag = (bf16*)(ws + OFF_ENCFRAG);
  float* hD   = (float*)(ws + OFF_ENCD);
  bf16* xh    = (bf16*)(ws + OFF_XH);

  // pre-swizzle Whh1/Wih1 chunk into WMMA B-fragment layout (resident all 2048 steps)
  for (int idx = tid; idx < 12*8*512; idx += NTHREADS) {
    int s = idx & 15, l2 = (idx >> 4) & 31, kt = (idx >> 9) & 7, nt = idx >> 12;
    int nn = l2 & 15, hf = l2 >> 4;
    int c = nt*16 + nn, g = c >> 6, u = c & 63;
    int j = g*Hz + chunk*64 + u;
    WfHH[idx] = (bf16)Whh1[(size_t)j*Hz + kt*32 + kmap(hf, s)];
  }
  for (int idx = tid; idx < 12*512; idx += NTHREADS) {
    int s = idx & 15, l2 = (idx >> 4) & 31, nt = idx >> 9;
    int nn = l2 & 15, hf = l2 >> 4;
    int c = nt*16 + nn, g = c >> 6, u = c & 63;
    int j = g*Hz + chunk*64 + u;
    WfIH[idx] = (bf16)Wih1[(size_t)j*Fz + kmap(hf, s)];
  }
  for (int c = tid; c < 192; c += NTHREADS) {
    int g = c >> 6, u = c & 63, col = g*Hz + chunk*64 + u;
    bI[c] = bih1[col]; bH[c] = bhh1[col];
  }
#if HAVE_TDM
  if (tid == 0) tdm_load_x_tile(x + (size_t)(Lz-1)*Fz, xstag);   // prefetch it=0
#endif
  __syncthreads();

  const int n = lane & 15, half = lane >> 4;
  unsigned bar = 0;
  for (int it = 0; it < Lz; ++it) {
    const int t = Lz-1-it;
    const int rp = it & 1, wp = rp ^ 1;
#if HAVE_TDM
    if (it+1 < Lz && tid == 0)
      tdm_load_x_tile(x + (size_t)(t-1)*Fz, xstag + ((it+1)&1)*(128*32));
#endif
    const v8f vzero = {0.f,0.f,0.f,0.f,0.f,0.f,0.f,0.f};
    v8f accRZ[8], accN[4], accNX[4];
#pragma unroll
    for (int i = 0; i < 8; i++) accRZ[i] = vzero;
#pragma unroll
    for (int i = 0; i < 4; i++) { accN[i] = vzero; accNX[i] = vzero; }

    const bf16* hin = hfrag + (size_t)rp*FRAG_ELEMS;
#pragma unroll 2
    for (int kt = 0; kt < 8; ++kt) {
      v16bf a = *(const v16bf*)(hin + ((size_t)((wave*8 + kt)*32 + lane))*16);
#pragma unroll
      for (int nt = 0; nt < 12; ++nt) {
        v16bf bm = *(const v16bf*)(WfHH + ((size_t)((nt*8 + kt)*32 + lane))*16);
        if (nt < 8) accRZ[nt]  = wmma_bf(a, bm, accRZ[nt]);
        else        accN[nt-8] = wmma_bf(a, bm, accN[nt-8]);
      }
    }
#if HAVE_TDM
    if (it+1 < Lz) { if (wave == 0) __builtin_amdgcn_s_wait_tensorcnt(1); }
    else           { if (wave == 0) __builtin_amdgcn_s_wait_tensorcnt(0); }
#else
    for (int idx = tid; idx < Bz*Fz; idx += NTHREADS) {
      int b = idx >> 5, f = idx & 31;
      xstag[(it&1)*(128*32) + idx] = x[(size_t)b*(Lz*Fz) + (size_t)t*Fz + f];
    }
#endif
    __syncthreads();
    {
      const float* xs = xstag + (it&1)*(128*32);
      v16bf a;
#pragma unroll
      for (int s = 0; s < 16; s++)
        a[s] = (bf16)xs[(wave*16 + n)*Fz + kmap(half, s)];
#pragma unroll
      for (int nt = 0; nt < 12; ++nt) {
        v16bf bm = *(const v16bf*)(WfIH + ((size_t)(nt*32 + lane))*16);
        if (nt < 8) accRZ[nt]   = wmma_bf(a, bm, accRZ[nt]);
        else        accNX[nt-8] = wmma_bf(a, bm, accNX[nt-8]);
      }
    }
    // gate combine + h update
    const float* hDr = hD + (size_t)rp*DLAY_ELEMS;
    float* hDw = hD + (size_t)wp*DLAY_ELEMS;
#pragma unroll
    for (int i = 0; i < 4; i++) {
      int cR = i*16 + n, cZ = cR + 64, cN = cR + 128;
#pragma unroll
      for (int r = 0; r < 8; r++) {
        float rg = 1.f/(1.f + __expf(-(accRZ[i][r]   + bI[cR] + bH[cR])));
        float zg = 1.f/(1.f + __expf(-(accRZ[4+i][r] + bI[cZ] + bH[cZ])));
        float ng = tanhf(accNX[i][r] + bI[cN] + rg*(accN[i][r] + bH[cN]));
        size_t doff = ((((size_t)chunk*8 + wave)*4 + i)*8 + r)*32 + lane;
        float hn = (1.f - zg)*ng + zg*hDr[doff];
        hDw[doff] = hn;
        int b = wave*16 + r + 8*half;
        stag[b*64 + i*16 + n] = (bf16)hn;
      }
    }
    __syncthreads();
    // bounce: frag layout + plain xh[t]
    bf16* hout = hfrag + (size_t)wp*FRAG_ELEMS;
#pragma unroll
    for (int rr = 0; rr < 2; rr++) {
      int row = tid*2 + rr;
      int l2 = row & 31, ktl = (row >> 5) & 1, mt = row >> 6;
      int mm = l2 & 15, hf = l2 >> 4;
      v16bf tv;
#pragma unroll
      for (int s = 0; s < 16; s++)
        tv[s] = stag[(mt*16 + mm)*64 + ktl*32 + kmap(hf, s)];
      *(v16bf*)(hout + ((size_t)((mt*8 + (chunk*2 + ktl))*32 + l2))*16) = tv;
    }
    {
      int b = tid >> 1, seg = tid & 1;
      const v16bf* src = (const v16bf*)(stag + b*64 + seg*32);
      v16bf* dst = (v16bf*)(xh + ((size_t)t*Bz + b)*Hz + chunk*64 + seg*32);
      dst[0] = src[0]; dst[1] = src[1];
    }
    bar += NWG;
    grid_barrier(ctr + 0, bar);
  }
}

// ---------------- decoder: forward scan ----------------
__global__ void __launch_bounds__(NTHREADS) dec_kernel(
    const float* __restrict__ x, const float* __restrict__ eps,
    const float* __restrict__ Wih2, const float* __restrict__ Whh2,
    const float* __restrict__ bih2, const float* __restrict__ bhh2,
    const float* __restrict__ Wih3, const float* __restrict__ Whh3,
    const float* __restrict__ bih3, const float* __restrict__ bhh3,
    const float* __restrict__ Wzl, const float* __restrict__ bzl,
    const float* __restrict__ Wzs, const float* __restrict__ bzs,
    const float* __restrict__ Wxl, const float* __restrict__ bxl,
    const float* __restrict__ Wxs, const float* __restrict__ bxs,
    float* __restrict__ out, char* __restrict__ ws)
{
  extern __shared__ char smem[];
  bf16* Wf2  = (bf16*)smem;                  // 12*8*512
  bf16* Wf3  = Wf2 + 12*8*512;               // 12*8*512
  bf16* Wfx  = Wf3 + 12*8*512;               // 8*512 (8 f-cols loc + 8 logvar)
  bf16* stag = Wfx + 8*512;                  // 128*64
  float* Wih2c = (float*)(stag + 128*64);    // 192*4
  float* Wih3c = Wih2c + 192*Zz;
  float* bI2 = Wih3c + 192*Zz; float* bH2 = bI2 + 192;
  float* bI3 = bH2 + 192;      float* bH3 = bI3 + 192;
  float* Wzc = bH3 + 192;                    // 8*512
  float* zbuf = Wzc + 8*512;                 // 128*4

  const int chunk = blockIdx.x, tid = threadIdx.x;
  const int wave = tid >> 5, lane = tid & 31;

  unsigned* ctr = (unsigned*)ws;
  float* partials = (float*)(ws + OFF_PART);
  bf16* phifrag = (bf16*)(ws + OFF_PHIFRAG);
  float* phiD   = (float*)(ws + OFF_PHID);
  bf16* phipl   = (bf16*)(ws + OFF_PHIPL);
  bf16* thfrag  = (bf16*)(ws + OFF_THFRAG);
  float* thD    = (float*)(ws + OFF_THD);
  const bf16* xh = (const bf16*)(ws + OFF_XH);

  for (int idx = tid; idx < 12*8*512; idx += NTHREADS) {
    int s = idx & 15, l2 = (idx >> 4) & 31, kt = (idx >> 9) & 7, nt = idx >> 12;
    int nn = l2 & 15, hf = l2 >> 4;
    int c = nt*16 + nn, g = c >> 6, u = c & 63;
    int j = g*Hz + chunk*64 + u;
    int k = kt*32 + kmap(hf, s);
    Wf2[idx] = (bf16)Whh2[(size_t)j*Hz + k];
    Wf3[idx] = (bf16)Whh3[(size_t)j*Hz + k];
  }
  for (int idx = tid; idx < 8*512; idx += NTHREADS) {
    int s = idx & 15, l2 = (idx >> 4) & 31, kt = idx >> 9;
    int nn = l2 & 15, hf = l2 >> 4;
    const float* W = (nn < 8) ? Wxl : Wxs;
    int j = chunk*8 + (nn & 7);
    Wfx[idx] = (bf16)W[(size_t)j*Hz + kt*32 + kmap(hf, s)];
  }
  for (int idx = tid; idx < 192*Zz; idx += NTHREADS) {
    int c = idx / Zz, k = idx % Zz;
    int g = c >> 6, u = c & 63, j = g*Hz + chunk*64 + u;
    Wih2c[idx] = Wih2[(size_t)j*Zz + k];
    Wih3c[idx] = Wih3[(size_t)j*Zz + k];
  }
  for (int c = tid; c < 192; c += NTHREADS) {
    int g = c >> 6, u = c & 63, col = g*Hz + chunk*64 + u;
    bI2[c] = bih2[col]; bH2[c] = bhh2[col];
    bI3[c] = bih3[col]; bH3[c] = bhh3[col];
  }
  for (int idx = tid; idx < 8*512; idx += NTHREADS) {
    int rrow = idx >> 9, j = idx & 511;
    Wzc[idx] = (rrow < 4) ? Wzl[(size_t)rrow*(2*Hz) + j] : Wzs[(size_t)(rrow-4)*(2*Hz) + j];
  }
  for (int idx = tid; idx < Bz*Zz; idx += NTHREADS) zbuf[idx] = 0.f;
  __syncthreads();

  int bb[2], zz2[2]; float bzl_r[2], bzs_r[2];
#pragma unroll
  for (int pi = 0; pi < 2; pi++) {
    int pair = tid*2 + pi; bb[pi] = pair >> 2; zz2[pi] = pair & 3;
    bzl_r[pi] = bzl[zz2[pi]]; bzs_r[pi] = bzs[zz2[pi]];
  }
  const int n = lane & 15, half = lane >> 4;
  const bool isLoc = (n < 8);
  const int fcol = chunk*8 + (n & 7);
  const float biasO = isLoc ? bxl[fcol] : bxs[fcol];

  float recon_acc = 0.f, kld_acc = 0.f;
  unsigned bar = 0;
  for (int t = 0; t < Lz; ++t) {
    const int rp = t & 1, wp = rp ^ 1;

    // phase 1: phi GRU (input = z_{t-1} via VALU, h via WMMA)
    gru_wmma_step(Wf2, Wih2c, bI2, bH2, zbuf,
                  phifrag + (size_t)rp*FRAG_ELEMS, phifrag + (size_t)wp*FRAG_ELEMS,
                  phiD + (size_t)rp*DLAY_ELEMS,   phiD + (size_t)wp*DLAY_ELEMS,
                  stag, phipl, chunk, wave, lane);
    bar += NWG; grid_barrier(ctr + 1, bar);

    // phase 2: z = loc + exp(0.5*logvar)*eps ; kld (WG0 only)
#pragma unroll
    for (int pi = 0; pi < 2; pi++) {
      int b = bb[pi], zi = zz2[pi];
      float loc = bzl_r[pi], lv = bzs_r[pi];
      const bf16* ph = phipl + (size_t)b*Hz;
      const bf16* xr = xh + ((size_t)t*Bz + b)*Hz;
      const float* wl = Wzc + zi*512;
      const float* wsg = Wzc + (4 + zi)*512;
      for (int j = 0; j < Hz; j++) {
        float pv = (float)ph[j];
        loc += pv*wl[j]; lv += pv*wsg[j];
      }
      for (int j = 0; j < Hz; j++) {
        float xv = (float)xr[j];
        loc += xv*wl[Hz + j]; lv += xv*wsg[Hz + j];
      }
      float e = eps[((size_t)t*Bz + b)*Zz + zi];
      zbuf[b*Zz + zi] = loc + __expf(0.5f*lv)*e;
      if (chunk == 0) kld_acc += 0.5f*(__expf(lv) + loc*loc - 1.f - lv);
    }
    __syncthreads();

    // phase 3: theta GRU (input = z_t)
    gru_wmma_step(Wf3, Wih3c, bI3, bH3, zbuf,
                  thfrag + (size_t)rp*FRAG_ELEMS, thfrag + (size_t)wp*FRAG_ELEMS,
                  thD + (size_t)rp*DLAY_ELEMS,   thD + (size_t)wp*DLAY_ELEMS,
                  stag, (bf16*)0, chunk, wave, lane);
    bar += NWG; grid_barrier(ctr + 1, bar);

    // phase 4: x_loc / x_logvar (WMMA) + recon + y store
    {
      v8f acc = {0.f,0.f,0.f,0.f,0.f,0.f,0.f,0.f};
      const bf16* thin = thfrag + (size_t)wp*FRAG_ELEMS;
#pragma unroll
      for (int kt = 0; kt < 8; kt++) {
        v16bf a  = *(const v16bf*)(thin + ((size_t)((wave*8 + kt)*32 + lane))*16);
        v16bf bm = *(const v16bf*)(Wfx + ((size_t)(kt*32 + lane))*16);
        acc = wmma_bf(a, bm, acc);
      }
#pragma unroll
      for (int r = 0; r < 8; r++) {
        float v = acc[r] + biasO;
        float other = __shfl(v, lane + (isLoc ? 8 : -8), 32);
        if (isLoc) {
          int b = wave*16 + r + 8*half;
          float lv = other;
          float xv = x[(size_t)b*(Lz*Fz) + (size_t)t*Fz + fcol];
          float d = (xv - v) * __expf(-0.5f*lv);
          recon_acc += 0.5f*d*d + 0.5f*lv + 0.5f*LOG2PI;
          out[((size_t)b*Lz + t)*Fz + fcol] = v;
        }
      }
    }
  }

  // deterministic reductions
  __syncthreads();
  float* red = (float*)stag;
  red[tid] = recon_acc; __syncthreads();
  for (int o = 128; o > 0; o >>= 1) { if (tid < o) red[tid] += red[tid + o]; __syncthreads(); }
  if (tid == 0) partials[chunk] = red[0];
  __syncthreads();
  red[tid] = kld_acc; __syncthreads();
  for (int o = 128; o > 0; o >>= 1) { if (tid < o) red[tid] += red[tid + o]; __syncthreads(); }
  if (tid == 0 && chunk == 0) partials[4] = red[0];
}

__global__ void fin_kernel(const float* __restrict__ partials, float* __restrict__ out) {
  out[(size_t)Bz*Lz*Fz]     = partials[0] + partials[1] + partials[2] + partials[3];
  out[(size_t)Bz*Lz*Fz + 1] = partials[4];
}

// ---------------- host entry ----------------
extern "C" void kernel_launch(void* const* d_in, const int* in_sizes, int n_in,
                              void* d_out, int out_size, void* d_ws, size_t ws_size,
                              hipStream_t stream)
{
  (void)in_sizes; (void)n_in; (void)out_size; (void)ws_size;
  const float* x    = (const float*)d_in[0];
  const float* eps  = (const float*)d_in[1];
  const float* Wih1 = (const float*)d_in[2];
  const float* Whh1 = (const float*)d_in[3];
  const float* bih1 = (const float*)d_in[4];
  const float* bhh1 = (const float*)d_in[5];
  const float* Wih2 = (const float*)d_in[6];
  const float* Whh2 = (const float*)d_in[7];
  const float* bih2 = (const float*)d_in[8];
  const float* bhh2 = (const float*)d_in[9];
  const float* Wih3 = (const float*)d_in[10];
  const float* Whh3 = (const float*)d_in[11];
  const float* bih3 = (const float*)d_in[12];
  const float* bhh3 = (const float*)d_in[13];
  const float* Wzl  = (const float*)d_in[14];
  const float* bzl  = (const float*)d_in[15];
  const float* Wzs  = (const float*)d_in[16];
  const float* bzs  = (const float*)d_in[17];
  const float* Wxl  = (const float*)d_in[18];
  const float* bxl  = (const float*)d_in[19];
  const float* Wxs  = (const float*)d_in[20];
  const float* bxs  = (const float*)d_in[21];

  (void)hipFuncSetAttribute((const void*)enc_kernel,
                      hipFuncAttributeMaxDynamicSharedMemorySize, ENC_LDS_BYTES);
  (void)hipFuncSetAttribute((const void*)dec_kernel,
                      hipFuncAttributeMaxDynamicSharedMemorySize, DEC_LDS_BYTES);

  (void)hipMemsetAsync(d_ws, 0, INIT_BYTES, stream);  // barrier ctrs + parity-0 state
  enc_kernel<<<NWG, NTHREADS, ENC_LDS_BYTES, stream>>>(
      x, Wih1, Whh1, bih1, bhh1, (char*)d_ws);
  dec_kernel<<<NWG, NTHREADS, DEC_LDS_BYTES, stream>>>(
      x, eps, Wih2, Whh2, bih2, bhh2, Wih3, Whh3, bih3, bhh3,
      Wzl, bzl, Wzs, bzs, Wxl, bxl, Wxs, bxs, (float*)d_out, (char*)d_ws);
  fin_kernel<<<1, 1, 0, stream>>>((const float*)((char*)d_ws + OFF_PART), (float*)d_out);
}
